// MultiHeadSelfAttention_58961311039834
// MI455X (gfx1250) — matrix operand order repaired and checked by reading, not compiled, not run on hardware
//
#include <hip/hip_runtime.h>
#include <hip/hip_bf16.h>

typedef __attribute__((ext_vector_type(16))) _Float16 v16h;
typedef __attribute__((ext_vector_type(8)))  _Float16 v8h;
typedef __attribute__((ext_vector_type(8)))  float    v8f;
typedef __attribute__((ext_vector_type(4)))  int      v4i;

#define BSZ   2
#define SEQ   2048
#define NDIM  1024
#define HEADS 16
#define HD    64
#define NEGV  (-1e9f)

// CDNA5 async global->LDS path (ASYNCcnt). Falls back to reg round-trip if the
// toolchain lacks the builtins (host pass always takes the fallback branch).
#if __has_builtin(__builtin_amdgcn_global_load_async_to_lds_b128) && \
    __has_builtin(__builtin_amdgcn_s_wait_asynccnt)
#define USE_ASYNC_LDS 1
typedef __attribute__((address_space(1))) v4i* g_b128_ptr;
typedef __attribute__((address_space(3))) v4i* l_b128_ptr;
#else
#define USE_ASYNC_LDS 0
#endif

__device__ __forceinline__ void stage16(const _Float16* g, _Float16* l) {
#if USE_ASYNC_LDS
  __builtin_amdgcn_global_load_async_to_lds_b128(
      (g_b128_ptr)(g), (l_b128_ptr)(l), /*imm offset*/ 0, /*cpol*/ 0);
#else
  *(v8h*)l = *(const v8h*)g;
#endif
}

__device__ __forceinline__ void async_wait_all() {
#if USE_ASYNC_LDS
  __builtin_amdgcn_s_wait_asynccnt(0);
#endif
}

__device__ __forceinline__ v16h join16(v8h lo, v8h hi) {
  return __builtin_shufflevector(lo, hi, 0,1,2,3,4,5,6,7,8,9,10,11,12,13,14,15);
}

// A-fragment (16 rows x 32 K, f16). ISA layout: lane&15 = row M; lanes<16 hold
// K[8h..8h+7] in v0-3 and K[16+8h..] in v4-7 (h = lane>>4): two 16B chunks.
__device__ __forceinline__ v16h load_afrag(const _Float16* base, int stride, int koff, int lane) {
  const int m  = lane & 15;
  const int hf = lane >> 4;
  const _Float16* r = base + m * stride + koff;
  v8h lo = *(const v8h*)(r + 8 * hf);
  v8h hi = *(const v8h*)(r + 16 + 8 * hf);
  return join16(lo, hi);
}

// B-fragment (32 K x 16 cols, f16) from a tile stored [n][k] row-major.
// lane&15 = col N; lane holds K[8h..8h+7] then K[16+8h..16+8h+7] (h = lane>>4).
__device__ __forceinline__ v16h load_bfrag(const _Float16* base, int stride, int koff, int lane) {
  const int n  = lane & 15;
  const int hf = lane >> 4;
  const _Float16* r = base + n * stride + koff;
  v8h lo = *(const v8h*)(r + 8 * hf);
  v8h hi = *(const v8h*)(r + 16 + 8 * hf);
  return join16(lo, hi);
}

__device__ __forceinline__ v8f wmma_f16(v16h a, v16h b, v8f c) {
  return __builtin_amdgcn_wmma_f32_16x16x32_f16(false, a, false, b, (short)0, c, false, false);
}

// ---------------------------------------------------------------------------
// f32 -> f16 conversion (8 elements / thread)
// ---------------------------------------------------------------------------
__global__ __launch_bounds__(256)
void cvt_f32_to_f16(const float* __restrict__ in, _Float16* __restrict__ out, int n) {
  int i = (blockIdx.x * blockDim.x + threadIdx.x) * 8;
  if (i + 8 <= n) {
    v8f a = *(const v8f*)(in + i);
    *(v8h*)(out + i) = __builtin_convertvector(a, v8h);
  }
}

// ---------------------------------------------------------------------------
// C = A[M x 1024] * Bw^T + bias.  Bw is [n][k] row-major f16 (i.e. W itself).
// Block tile 128x128, 8 waves (2x4 of 16x16 tiles each => 32x64 per wave).
// Double-buffered LDS filled via async-to-LDS; one barrier per K-step.
// mode: 0 = Q  ([b,h,s,d] f16, scaled)   1 = K ([b,h,s,d] f16)
//       2 = Vt ([b,h,d,s] f16)           3 = f32 [m][n] out
// ---------------------------------------------------------------------------
__global__ __launch_bounds__(256)
void gemm_proj(const _Float16* __restrict__ A, const _Float16* __restrict__ Bw,
               const float* __restrict__ bias, void* __restrict__ Cout,
               int mode, float scale) {
  __shared__ __align__(16) _Float16 As[2][128 * 40];  // padded stride: 40 halves
  __shared__ __align__(16) _Float16 Bs[2][128 * 40];

  const int tid   = threadIdx.x;
  const int lane  = tid & 31;
  const int w     = tid >> 5;
  const int waveM = w >> 1;      // 0..3
  const int waveN = w & 1;       // 0..1
  const int bm    = blockIdx.y * 128;
  const int bn    = blockIdx.x * 128;

  v8f zero = {};
  v8f acc[2][4];
#pragma unroll
  for (int mt = 0; mt < 2; ++mt)
#pragma unroll
    for (int nt = 0; nt < 4; ++nt) acc[mt][nt] = zero;

  const int srow   = tid >> 1;          // 0..127
  const int schunk = (tid & 1) * 16;    // 0 or 16 halves

  auto stage = [&](int buf, int k0) {
    const _Float16* ga = A  + (size_t)(bm + srow) * NDIM + k0 + schunk;
    const _Float16* gb = Bw + (size_t)(bn + srow) * NDIM + k0 + schunk;
    _Float16* la = &As[buf][srow * 40 + schunk];
    _Float16* lb = &Bs[buf][srow * 40 + schunk];
    stage16(ga,     la);
    stage16(ga + 8, la + 8);
    stage16(gb,     lb);
    stage16(gb + 8, lb + 8);
  };

  stage(0, 0);                          // prologue fill

  for (int k0 = 0; k0 < NDIM; k0 += 32) {
    const int buf = (k0 >> 5) & 1;
    async_wait_all();                   // this wave's fills for `buf` landed
    __syncthreads();                    // everyone's fills landed / reads done
    if (k0 + 32 < NDIM) stage(buf ^ 1, k0 + 32);  // overlap with compute

    v16h af[2], bf[4];
#pragma unroll
    for (int mt = 0; mt < 2; ++mt)
      af[mt] = load_afrag(&As[buf][(waveM * 32 + mt * 16) * 40], 40, 0, lane);
#pragma unroll
    for (int nt = 0; nt < 4; ++nt)
      bf[nt] = load_bfrag(&Bs[buf][(waveN * 64 + nt * 16) * 40], 40, 0, lane);
#pragma unroll
    for (int mt = 0; mt < 2; ++mt)
#pragma unroll
      for (int nt = 0; nt < 4; ++nt)
        acc[mt][nt] = wmma_f16(af[mt], bf[nt], acc[mt][nt]);
  }

  const int hf = lane >> 4;
  const int nl = lane & 15;
#pragma unroll
  for (int nt = 0; nt < 4; ++nt) {
    const int gn = bn + waveN * 64 + nt * 16 + nl;
    const float bv = bias[gn];
#pragma unroll
    for (int mt = 0; mt < 2; ++mt) {
#pragma unroll
      for (int r = 0; r < 8; ++r) {
        const int gm = bm + waveM * 32 + mt * 16 + r + 8 * hf;
        const float v = (acc[mt][nt][r] + bv) * scale;
        if (mode == 3) {
          ((float*)Cout)[(size_t)gm * NDIM + gn] = v;
        } else {
          const int bb = gm >> 11, s = gm & (SEQ - 1);
          const int hh = gn >> 6,  d = gn & (HD - 1);
          const _Float16 hv = (_Float16)v;
          if (mode == 2)  // V transposed: [b,h,d,s]
            ((_Float16*)Cout)[(((size_t)(bb * HEADS + hh)) * HD + d) * SEQ + s] = hv;
          else            // Q/K: [b,h,s,d]
            ((_Float16*)Cout)[(((size_t)(bb * HEADS + hh)) * SEQ + s) * HD + d] = hv;
        }
      }
    }
  }
}

// ---------------------------------------------------------------------------
// Flash attention. Block = (b, h, 128 q rows); 8 waves each own 16 q rows.
// K/V tiles of 64 keys double-buffered in LDS (async-to-LDS fills) and shared
// by all waves. Online softmax; Q pre-scaled by 1/sqrt(HD); V supplied
// transposed [b,h,d,s] so its B-fragments are contiguous in K (= key index).
// ---------------------------------------------------------------------------
__global__ __launch_bounds__(256)
void attn_kernel(const _Float16* __restrict__ Q, const _Float16* __restrict__ K,
                 const _Float16* __restrict__ Vt, const int* __restrict__ pad,
                 _Float16* __restrict__ Oh) {
  __shared__ __align__(16) _Float16 Ks[2][64 * 72];   // [key][d], padded
  __shared__ __align__(16) _Float16 Vs[2][64 * 72];   // [d][key], padded
  __shared__ __align__(16) _Float16 Ps[8 * 16 * 72];  // per-wave P tiles

  const int tid  = threadIdx.x;
  const int lane = tid & 31;
  const int w    = tid >> 5;
  const int lm   = lane & 15;
  const int hf   = lane >> 4;

  const int qb = blockIdx.x;   // 0..15
  const int h  = blockIdx.y;
  const int b  = blockIdx.z;

  const size_t headOff = ((size_t)(b * HEADS + h)) * SEQ * HD;
  const _Float16* Qg = Q  + headOff;
  const _Float16* Kg = K  + headOff;
  const _Float16* Vg = Vt + headOff;   // [d][s], row stride SEQ

  const int qrow = qb * 128 + w * 16;  // wave's first q row

  // Q fragments (registers for the whole kernel): A-layout, K = d.
  v16h qf[2];
#pragma unroll
  for (int s = 0; s < 2; ++s) {
    const _Float16* r = Qg + (size_t)(qrow + lm) * HD + 32 * s;
    v8h lo = *(const v8h*)(r + 8 * hf);
    v8h hi = *(const v8h*)(r + 16 + 8 * hf);
    qf[s] = join16(lo, hi);
  }

  v8f zero = {};
  v8f o[4];
#pragma unroll
  for (int dt = 0; dt < 4; ++dt) o[dt] = zero;
  float m_[8], l_[8];
#pragma unroll
  for (int r = 0; r < 8; ++r) { m_[r] = -INFINITY; l_[r] = 0.f; }

  const int srow6 = tid >> 2;        // 0..63 staging row
  const int sch   = (tid & 3) * 16;  // halves within row

  auto stageKV = [&](int buf, int kbase) {
    const _Float16* gk = Kg + (size_t)(kbase + srow6) * HD + sch;
    const _Float16* gv = Vg + (size_t)srow6 * SEQ + kbase + sch;
    _Float16* lk = &Ks[buf][srow6 * 72 + sch];
    _Float16* lv = &Vs[buf][srow6 * 72 + sch];
    stage16(gk,     lk);
    stage16(gk + 8, lk + 8);
    stage16(gv,     lv);
    stage16(gv + 8, lv + 8);
  };

  const int nkt = 2 * qb + 2;        // causal: keys up to this block's last q
  stageKV(0, 0);                     // prologue fill

  for (int kt = 0; kt < nkt; ++kt) {
    const int buf = kt & 1;
    async_wait_all();
    __syncthreads();                 // current tile visible; old buffer free
    if (kt + 1 < nkt) stageKV(buf ^ 1, (kt + 1) * 64);

    const int kbase = kt * 64;

    // S = Q * K^T  (16 x 64, K-dim = HD = 64 -> 2 WMMA k-steps)
    v8f sacc[4];
#pragma unroll
    for (int nt = 0; nt < 4; ++nt) sacc[nt] = zero;
#pragma unroll
    for (int s = 0; s < 2; ++s)
#pragma unroll
      for (int nt = 0; nt < 4; ++nt) {
        v16h bf = load_bfrag(&Ks[buf][nt * 16 * 72], 72, 32 * s, lane);
        sacc[nt] = wmma_f16(qf[s], bf, sacc[nt]);
      }

    // masks: causal (k <= q) + padding
#pragma unroll
    for (int nt = 0; nt < 4; ++nt) {
      const int kg = kbase + nt * 16 + lm;
      const bool kvalid = (pad[b * SEQ + kg] != 0);
#pragma unroll
      for (int r = 0; r < 8; ++r) {
        const int qg = qrow + r + 8 * hf;
        float sv = sacc[nt][r];
        if (!kvalid || kg > qg) sv = NEGV;
        sacc[nt][r] = sv;
      }
    }

    // online softmax update (row stats replicated across each 16-lane group)
    float alpha[8];
#pragma unroll
    for (int r = 0; r < 8; ++r) {
      float mx = fmaxf(fmaxf(sacc[0][r], sacc[1][r]), fmaxf(sacc[2][r], sacc[3][r]));
#pragma unroll
      for (int d = 1; d < 16; d <<= 1) mx = fmaxf(mx, __shfl_xor(mx, d, 32));
      const float mn = fmaxf(m_[r], mx);
      alpha[r] = __expf(m_[r] - mn);
      float rs = 0.f;
#pragma unroll
      for (int nt = 0; nt < 4; ++nt) {
        const float p = __expf(sacc[nt][r] - mn);
        sacc[nt][r] = p;
        rs += p;
      }
#pragma unroll
      for (int d = 1; d < 16; d <<= 1) rs += __shfl_xor(rs, d, 32);
      l_[r] = l_[r] * alpha[r] + rs;
      m_[r] = mn;
    }
#pragma unroll
    for (int dt = 0; dt < 4; ++dt)
#pragma unroll
      for (int r = 0; r < 8; ++r) o[dt][r] *= alpha[r];

    // P: C-layout (lane = col) -> LDS -> A-layout (lane = row), f16
    _Float16* pw = Ps + w * (16 * 72);
#pragma unroll
    for (int nt = 0; nt < 4; ++nt)
#pragma unroll
      for (int r = 0; r < 8; ++r)
        pw[(r + 8 * hf) * 72 + nt * 16 + lm] = (_Float16)sacc[nt][r];

    // O += P * V   (K-dim = 64 keys -> 2 WMMA k-steps)
#pragma unroll
    for (int s = 0; s < 2; ++s) {
      v16h pf = load_afrag(pw, 72, 32 * s, lane);
#pragma unroll
      for (int dt = 0; dt < 4; ++dt) {
        v16h vf = load_bfrag(&Vs[buf][dt * 16 * 72], 72, 32 * s, lane);
        o[dt] = wmma_f16(pf, vf, o[dt]);
      }
    }
  }

  // epilogue: normalize and store [b, s, h*64 + d] as f16
#pragma unroll
  for (int r = 0; r < 8; ++r) {
    const float inv = 1.0f / l_[r];
    const int qg = qrow + r + 8 * hf;
    const size_t rowbase = ((size_t)(b * SEQ + qg)) * NDIM + h * HD;
#pragma unroll
    for (int dt = 0; dt < 4; ++dt)
      Oh[rowbase + dt * 16 + lm] = (_Float16)(o[dt][r] * inv);
  }
}

// ---------------------------------------------------------------------------
extern "C" void kernel_launch(void* const* d_in, const int* in_sizes, int n_in,
                              void* d_out, int out_size, void* d_ws, size_t ws_size,
                              hipStream_t stream) {
  (void)in_sizes; (void)n_in; (void)out_size; (void)ws_size;
  const float* x  = (const float*)d_in[0];
  const int*  pad = (const int*)d_in[1];
  // d_in[2] = causal mask (tril) -- realized arithmetically as k <= q
  const float* Wq = (const float*)d_in[3];
  const float* bq = (const float*)d_in[4];
  const float* Wk = (const float*)d_in[5];
  const float* bk = (const float*)d_in[6];
  const float* Wv = (const float*)d_in[7];
  const float* bv = (const float*)d_in[8];
  const float* Wo = (const float*)d_in[9];
  const float* bo = (const float*)d_in[10];

  const size_t NX = (size_t)BSZ * SEQ * NDIM;  // 4,194,304
  const size_t NW = (size_t)NDIM * NDIM;       // 1,048,576

  _Float16* Xh  = (_Float16*)d_ws;
  _Float16* Wqh = Xh  + NX;
  _Float16* Wkh = Wqh + NW;
  _Float16* Wvh = Wkh + NW;
  _Float16* Woh = Wvh + NW;
  _Float16* Qd  = Woh + NW;
  _Float16* Kd  = Qd  + NX;
  _Float16* Vtd = Kd  + NX;
  _Float16* Ohd = Vtd + NX;   // total ~48 MB of f16 workspace

  cvt_f32_to_f16<<<(int)(NX / (8 * 256)), 256, 0, stream>>>(x,  Xh,  (int)NX);
  cvt_f32_to_f16<<<(int)(NW / (8 * 256)), 256, 0, stream>>>(Wq, Wqh, (int)NW);
  cvt_f32_to_f16<<<(int)(NW / (8 * 256)), 256, 0, stream>>>(Wk, Wkh, (int)NW);
  cvt_f32_to_f16<<<(int)(NW / (8 * 256)), 256, 0, stream>>>(Wv, Wvh, (int)NW);
  cvt_f32_to_f16<<<(int)(NW / (8 * 256)), 256, 0, stream>>>(Wo, Woh, (int)NW);

  dim3 ggrid(NDIM / 128, (BSZ * SEQ) / 128);   // 8 x 32
  gemm_proj<<<ggrid, 256, 0, stream>>>(Xh, Wqh, bq, Qd,  0, 0.125f);  // 1/sqrt(64)
  gemm_proj<<<ggrid, 256, 0, stream>>>(Xh, Wkh, bk, Kd,  1, 1.0f);
  gemm_proj<<<ggrid, 256, 0, stream>>>(Xh, Wvh, bv, Vtd, 2, 1.0f);

  dim3 agrid(SEQ / 128, HEADS, BSZ);           // 16 x 16 x 2
  attn_kernel<<<agrid, 256, 0, stream>>>(Qd, Kd, Vtd, pad, Ohd);

  gemm_proj<<<ggrid, 256, 0, stream>>>(Ohd, Woh, bo, (float*)d_out, 3, 1.0f);
}
